// LLaMABlock_33998961115812
// MI455X (gfx1250) — matrix operand order repaired and checked by reading, not compiled, not run on hardware
//
#include <hip/hip_runtime.h>
#include <hip/hip_bf16.h>

// ---------------------------------------------------------------------------
// LLaMA block on gfx1250 via v_wmma_f32_16x16x32_f16 (mixed precision:
// f16 operands, f32 accumulate). All GEMMs + attention use WMMA.
// GEMM: one wave computes a 32x64 register tile (2 A-frags x 4 B-frags,
// 8 accumulators) for ~4x A-reuse / 2x B-reuse vs a single 16x16 tile.
// ---------------------------------------------------------------------------

typedef __attribute__((ext_vector_type(16))) _Float16 v16h;
typedef __attribute__((ext_vector_type(8)))  _Float16 v8h;
typedef __attribute__((ext_vector_type(8)))  float    v8f;

#define DIM    512
#define NTOK   8192      // B*N = 4*2048
#define SEQ    2048
#define NH     8
#define HD     64
#define HID    1365
#define HIDP   1408      // padded to multiple of 64 (22 * 64)

// ---- WMMA helpers ---------------------------------------------------------

__device__ __forceinline__ v8f wmma_f16(v16h a, v16h b, v8f c) {
  // D = A(16x32 f16) * B(32x16 f16) + C(16x16 f32)
  return __builtin_amdgcn_wmma_f32_16x16x32_f16(
      /*neg_a=*/false, a, /*neg_b=*/false, b,
      /*c_mod=*/(short)0, c, /*reuse_a=*/false, /*reuse_b=*/false);
}

// A-fragment (16x32, f16) from row-major [M x lda] matrix.
// ISA layout: lane l -> row m = l&15; K halves = base+{0..7}, base+16+{0..7},
// base = 8 if lane>=16 else 0. Two contiguous 16B loads.
__device__ __forceinline__ v16h load_a_frag(const _Float16* A, int lda,
                                            int m0, int k0) {
  int lane = threadIdx.x & 31;
  int m  = m0 + (lane & 15);
  int kb = k0 + ((lane & 16) ? 8 : 0);
  const _Float16* p = A + (size_t)m * lda + kb;
  v16h r;
  *(v8h*)&r        = *(const v8h*)p;
  *((v8h*)&r + 1)  = *(const v8h*)(p + 16);
  return r;
}

// B-fragment (32x16, f16) from K-major [K x ldb] matrix (i.e. W^T).
// ISA layout: lane l -> column n = n0+(l&15); K = base+{0..7}, base+16+{0..7},
// base = 8 if lane>=16 else 0. Strided scalar loads.
__device__ __forceinline__ v16h load_b_frag(const _Float16* BT, int ldb,
                                            int k0, int n0) {
  int lane = threadIdx.x & 31;
  int kb = k0 + ((lane & 16) ? 8 : 0);
  const _Float16* p = BT + (size_t)kb * ldb + n0 + (lane & 15);
  v16h r;
#pragma unroll
  for (int i = 0; i < 8; ++i) {
    r[i]     = p[(size_t)i * ldb];
    r[8 + i] = p[(size_t)(16 + i) * ldb];
  }
  return r;
}

// ---- weight conversion: f32 [N x K] row-major -> f16 W^T [Kd x Nd], padded 0
__global__ void transpose_f16_kernel(const float* __restrict__ src,
                                     _Float16* __restrict__ dst,
                                     int N, int K, int Nd, int Kd) {
  int i = blockIdx.x * blockDim.x + threadIdx.x;
  int total = Kd * Nd;
  if (i >= total) return;
  int k = i / Nd;
  int n = i - k * Nd;
  dst[i] = (k < K && n < N) ? (_Float16)src[(size_t)n * K + k] : (_Float16)0.0f;
}

// ---- rmsnorm: f32 [rows x 512] -> f16 [rows x 512] ------------------------
__global__ __launch_bounds__(256) void rmsnorm_kernel(
    const float* __restrict__ x, const float* __restrict__ w,
    _Float16* __restrict__ out) {
  int row = blockIdx.x;
  int tid = threadIdx.x;
  const float* xr = x + (size_t)row * DIM;
  float v0 = xr[tid];
  float v1 = xr[tid + 256];
  float ss = v0 * v0 + v1 * v1;
#pragma unroll
  for (int off = 1; off < 32; off <<= 1) ss += __shfl_xor(ss, off, 32);
  __shared__ float part[8];
  if ((tid & 31) == 0) part[tid >> 5] = ss;
  __syncthreads();
  float tot = part[0] + part[1] + part[2] + part[3] +
              part[4] + part[5] + part[6] + part[7];
  float inv = rsqrtf(tot * (1.0f / DIM) + 1e-6f);
  out[(size_t)row * DIM + tid]       = (_Float16)(v0 * inv * w[tid]);
  out[(size_t)row * DIM + tid + 256] = (_Float16)(v1 * inv * w[tid + 256]);
}

// ---- generic WMMA GEMM: out[M x N] = A[M x K](f16) * BT[K x N](f16) -------
// one wave computes a 32x64 register tile; optional bias / residual / f16 out.
__global__ __launch_bounds__(256) void gemm_wmma_kernel(
    const _Float16* __restrict__ A, const _Float16* __restrict__ BT,
    const float* __restrict__ bias, const float* __restrict__ residual,
    float* __restrict__ outf, _Float16* __restrict__ outh,
    int M, int N, int K) {
  int wave   = (blockIdx.x * blockDim.x + threadIdx.x) >> 5;
  int tilesN = N >> 6;                 // 64-wide N tiles
  int tm = wave / tilesN;
  int tn = wave - tm * tilesN;
  if (tm * 32 >= M) return;
  int m0 = tm * 32, n0 = tn * 64;

  v8f acc[2][4];
#pragma unroll
  for (int i = 0; i < 2; ++i)
#pragma unroll
    for (int j = 0; j < 4; ++j) acc[i][j] = v8f{};

  for (int k0 = 0; k0 < K; k0 += 32) {
    v16h a0 = load_a_frag(A, K, m0,      k0);
    v16h a1 = load_a_frag(A, K, m0 + 16, k0);
#pragma unroll
    for (int j = 0; j < 4; ++j) {
      v16h b = load_b_frag(BT, N, k0, n0 + j * 16);
      acc[0][j] = wmma_f16(a0, b, acc[0][j]);
      acc[1][j] = wmma_f16(a1, b, acc[1][j]);
    }
  }

  int lane = threadIdx.x & 31;
  int nc = lane & 15;
  int mh = (lane & 16) ? 8 : 0;
#pragma unroll
  for (int i = 0; i < 2; ++i)
#pragma unroll
    for (int j = 0; j < 4; ++j) {
      int n = n0 + j * 16 + nc;
      float bv = bias ? bias[n] : 0.0f;
#pragma unroll
      for (int r = 0; r < 8; ++r) {
        int m = m0 + i * 16 + r + mh;
        size_t idx = (size_t)m * N + n;
        float o = acc[i][j][r] + bv;
        if (residual) o += residual[idx];
        if (outh) outh[idx] = (_Float16)o;
        else      outf[idx] = o;
      }
    }
}

// ---- qkv scatter: f16 [tok x 1536] -> Q[B,H,N,HD], Kt[B,H,HD,N], V[B,H,N,HD]
__global__ void qkv_scatter_kernel(const _Float16* __restrict__ qkv,
                                   _Float16* __restrict__ Qo,
                                   _Float16* __restrict__ Kt,
                                   _Float16* __restrict__ Vo) {
  int i = blockIdx.x * blockDim.x + threadIdx.x;
  if (i >= NTOK * 3 * DIM) return;
  int token = i / (3 * DIM);
  int c     = i - token * (3 * DIM);
  int b = token >> 11, t = token & (SEQ - 1);
  int sel = c / DIM, rem = c - sel * DIM;
  int h = rem >> 6, d = rem & 63;
  _Float16 v = qkv[i];
  size_t bh = (size_t)(b * NH + h);
  if (sel == 0)      Qo[(bh * SEQ + t) * HD + d] = v;
  else if (sel == 1) Kt[(bh * HD + d) * SEQ + t] = v;
  else               Vo[(bh * SEQ + t) * HD + d] = v;
}

// ---- attention: one wave per 16 query rows, two-pass softmax, all WMMA ----
__global__ __launch_bounds__(32) void attn_kernel(
    const _Float16* __restrict__ Q, const _Float16* __restrict__ Kt,
    const _Float16* __restrict__ V, _Float16* __restrict__ O) {
  int qt = blockIdx.x & 127;          // query tile (N/16 = 128)
  int h  = (blockIdx.x >> 7) & 7;
  int b  = blockIdx.x >> 10;
  int lane = threadIdx.x & 31;
  size_t bh = (size_t)(b * NH + h);
  const _Float16* Qb  = Q  + bh * SEQ * HD;
  const _Float16* Ktb = Kt + bh * HD * SEQ;
  const _Float16* Vb  = V  + bh * SEQ * HD;
  int q0 = qt * 16;
  const float scale = 0.125f;         // HD^-0.5

  // Q fragments held in registers for both passes
  v16h qa0 = load_a_frag(Qb, HD, q0, 0);
  v16h qa1 = load_a_frag(Qb, HD, q0, 32);

  int mh = (lane & 16) ? 8 : 0;
  int nc = lane & 15;

  // pass 1: per-row max of scaled scores
  float rm[8];
#pragma unroll
  for (int r = 0; r < 8; ++r) rm[r] = -1e30f;
  for (int kk = 0; kk < SEQ; kk += 16) {
    v8f s = {};
    s = wmma_f16(qa0, load_b_frag(Ktb, SEQ, 0,  kk), s);
    s = wmma_f16(qa1, load_b_frag(Ktb, SEQ, 32, kk), s);
#pragma unroll
    for (int r = 0; r < 8; ++r) {
      float v = s[r] * scale;
      v = fmaxf(v, __shfl_xor(v, 1, 32));
      v = fmaxf(v, __shfl_xor(v, 2, 32));
      v = fmaxf(v, __shfl_xor(v, 4, 32));
      v = fmaxf(v, __shfl_xor(v, 8, 32));
      rm[r] = fmaxf(rm[r], v);
    }
  }

  // pass 2: P = exp(S - max); O += P*V; rowsum accumulate
  __shared__ __align__(32) _Float16 Plds[16 * 32];
  float rs[8];
#pragma unroll
  for (int r = 0; r < 8; ++r) rs[r] = 0.0f;
  v8f acc[4] = {v8f{}, v8f{}, v8f{}, v8f{}};

  for (int kk = 0; kk < SEQ; kk += 32) {
    v8f s0 = {}, s1 = {};
    s0 = wmma_f16(qa0, load_b_frag(Ktb, SEQ, 0,  kk), s0);
    s0 = wmma_f16(qa1, load_b_frag(Ktb, SEQ, 32, kk), s0);
    s1 = wmma_f16(qa0, load_b_frag(Ktb, SEQ, 0,  kk + 16), s1);
    s1 = wmma_f16(qa1, load_b_frag(Ktb, SEQ, 32, kk + 16), s1);
#pragma unroll
    for (int r = 0; r < 8; ++r) {
      float p0 = __expf(s0[r] * scale - rm[r]);
      float p1 = __expf(s1[r] * scale - rm[r]);
      int m = r + mh;
      Plds[m * 32 + nc]      = (_Float16)p0;
      Plds[m * 32 + 16 + nc] = (_Float16)p1;
      float v = p0 + p1;
      v += __shfl_xor(v, 1, 32);
      v += __shfl_xor(v, 2, 32);
      v += __shfl_xor(v, 4, 32);
      v += __shfl_xor(v, 8, 32);
      rs[r] += v;
    }
    __syncthreads();                       // P visible for A-frag load
    v16h pa = load_a_frag(Plds, 32, 0, 0); // layout-convert via LDS
#pragma unroll
    for (int d = 0; d < 4; ++d)
      acc[d] = wmma_f16(pa, load_b_frag(Vb, HD, kk, d * 16), acc[d]);
    __syncthreads();                       // done reading before next write
  }

  // epilogue: normalize and store f16 attention output [tok x 512]
#pragma unroll
  for (int d = 0; d < 4; ++d)
#pragma unroll
    for (int r = 0; r < 8; ++r) {
      int m = r + mh;
      float o = acc[d][r] / rs[r];
      size_t tok = (size_t)b * SEQ + q0 + m;
      O[tok * DIM + h * HD + d * 16 + nc] = (_Float16)o;
    }
}

// ---- silu(h1) * h3, f16 in/out --------------------------------------------
__global__ void silu_mul_kernel(const _Float16* __restrict__ h1,
                                const _Float16* __restrict__ h3,
                                _Float16* __restrict__ a, int total) {
  int i = blockIdx.x * blockDim.x + threadIdx.x;
  if (i >= total) return;
  float v = (float)h1[i];
  float s = v / (1.0f + __expf(-v));
  a[i] = (_Float16)(s * (float)h3[i]);
}

// ---------------------------------------------------------------------------
extern "C" void kernel_launch(void* const* d_in, const int* in_sizes, int n_in,
                              void* d_out, int out_size, void* d_ws, size_t ws_size,
                              hipStream_t stream) {
  const float* x      = (const float*)d_in[0];
  const float* n1_w   = (const float*)d_in[1];
  const float* n2_w   = (const float*)d_in[2];
  const float* qkv_w  = (const float*)d_in[3];
  const float* qkv_b  = (const float*)d_in[4];
  const float* proj_w = (const float*)d_in[5];
  const float* proj_b = (const float*)d_in[6];
  const float* w1     = (const float*)d_in[7];
  const float* w2     = (const float*)d_in[8];
  const float* w3     = (const float*)d_in[9];
  float* out = (float*)d_out;

  // ---- workspace layout (overlapped; peak ~85 MB) ----
  char* base = (char*)d_ws;
  size_t off = 0;
  auto take = [&](size_t bytes) -> char* {
    char* p = base + off;
    off = (off + bytes + 255) & ~(size_t)255;
    return p;
  };
  _Float16* qkv_wT  = (_Float16*)take((size_t)DIM * 3 * DIM * 2);   // [512 x 1536]
  _Float16* proj_wT = (_Float16*)take((size_t)DIM * DIM * 2);       // [512 x 512]
  _Float16* w1T     = (_Float16*)take((size_t)DIM * HIDP * 2);      // [512 x 1408]
  _Float16* w3T     = (_Float16*)take((size_t)DIM * HIDP * 2);
  _Float16* w2T     = (_Float16*)take((size_t)HIDP * DIM * 2);      // [1408 x 512]
  _Float16* actA    = (_Float16*)take((size_t)NTOK * DIM * 2);      // h / attn / g
  // regionB: qkv f16 (24 MB), later h1 + h3 f16 (2 x 23.1 MB)
  char* regionB = take((size_t)2 * NTOK * HIDP * 2);
  // regionC: Q,Kt,V f16 (3 x 8.4 MB), later a_f16 (23.1 MB)
  char* regionC = take((size_t)3 * NTOK * HD * NH * 2);
  _Float16* qkv_f16 = (_Float16*)regionB;
  _Float16* h1      = (_Float16*)regionB;
  _Float16* h3      = (_Float16*)(regionB + (size_t)NTOK * HIDP * 2);
  _Float16* Qb = (_Float16*)regionC;
  _Float16* Kt = Qb + (size_t)NTOK * HD;     // [B,H,HD,N]
  _Float16* Vb = Kt + (size_t)NTOK * HD;
  _Float16* aff = (_Float16*)regionC;        // reused after attention+proj

  // 1) weight conversion to f16 transposed (padded with zeros)
  {
    int t;
    t = DIM * 3 * DIM;
    transpose_f16_kernel<<<(t + 255) / 256, 256, 0, stream>>>(qkv_w, qkv_wT, 3 * DIM, DIM, 3 * DIM, DIM);
    t = DIM * DIM;
    transpose_f16_kernel<<<(t + 255) / 256, 256, 0, stream>>>(proj_w, proj_wT, DIM, DIM, DIM, DIM);
    t = DIM * HIDP;
    transpose_f16_kernel<<<(t + 255) / 256, 256, 0, stream>>>(w1, w1T, HID, DIM, HIDP, DIM);
    transpose_f16_kernel<<<(t + 255) / 256, 256, 0, stream>>>(w3, w3T, HID, DIM, HIDP, DIM);
    t = HIDP * DIM;
    transpose_f16_kernel<<<(t + 255) / 256, 256, 0, stream>>>(w2, w2T, DIM, HID, DIM, HIDP);
  }

  // 2) h = rmsnorm(x, n1_w) -> f16
  rmsnorm_kernel<<<NTOK, 256, 0, stream>>>(x, n1_w, actA);

  // 3) qkv = h @ qkv_w^T + qkv_b  (f16 out)
  {
    int waves = (NTOK / 32) * (3 * DIM / 64);
    gemm_wmma_kernel<<<(waves + 7) / 8, 256, 0, stream>>>(
        actA, qkv_wT, qkv_b, nullptr, nullptr, qkv_f16, NTOK, 3 * DIM, DIM);
  }

  // 4) scatter to Q [B,H,N,HD], K^T [B,H,HD,N], V [B,H,N,HD] in f16
  {
    int t = NTOK * 3 * DIM;
    qkv_scatter_kernel<<<(t + 255) / 256, 256, 0, stream>>>(qkv_f16, Qb, Kt, Vb);
  }

  // 5) attention -> actA (f16 [tok x 512])
  attn_kernel<<<4 * NH * (SEQ / 16), 32, 0, stream>>>(Qb, Kt, Vb, actA);

  // 6) x1 = x + attn @ proj_w^T + proj_b  -> d_out (f32)
  {
    int waves = (NTOK / 32) * (DIM / 64);
    gemm_wmma_kernel<<<(waves + 7) / 8, 256, 0, stream>>>(
        actA, proj_wT, proj_b, x, out, nullptr, NTOK, DIM, DIM);
  }

  // 7) g = rmsnorm(x1, n2_w) -> f16 (actA reused)
  rmsnorm_kernel<<<NTOK, 256, 0, stream>>>(out, n2_w, actA);

  // 8/9) h1 = g @ w1^T, h3 = g @ w3^T (f16 out, padded cols are exact zeros)
  {
    int waves = (NTOK / 32) * (HIDP / 64);
    gemm_wmma_kernel<<<(waves + 7) / 8, 256, 0, stream>>>(
        actA, w1T, nullptr, nullptr, nullptr, h1, NTOK, HIDP, DIM);
    gemm_wmma_kernel<<<(waves + 7) / 8, 256, 0, stream>>>(
        actA, w3T, nullptr, nullptr, nullptr, h3, NTOK, HIDP, DIM);
  }

  // 10) a = silu(h1) * h3 (f16)
  {
    int t = NTOK * HIDP;
    silu_mul_kernel<<<(t + 255) / 256, 256, 0, stream>>>(h1, h3, aff, t);
  }

  // 11) out = x1 + a @ w2^T  (K = 1408, padded rows contribute 0)
  {
    int waves = (NTOK / 32) * (DIM / 64);
    gemm_wmma_kernel<<<(waves + 7) / 8, 256, 0, stream>>>(
        aff, w2T, nullptr, out, out, nullptr, NTOK, DIM, HIDP);
  }
}
